// GNN_64647847740010
// MI455X (gfx1250) — matrix operand order repaired and checked by reading, not compiled, not run on hardware
//
#include <hip/hip_runtime.h>
#include <hip/hip_bf16.h>
#include <math.h>

#define Bn 128
#define Nn 512
#define Hn 128
#define Tn 4
#define LDP 136   // padded LDS row stride (bf16 elems): 272B -> 16B aligned, avoids 32-way bank conflicts

typedef __attribute__((ext_vector_type(16))) __bf16 bf16x16;
typedef __attribute__((ext_vector_type(8)))  __bf16 bf16x8;
typedef __attribute__((ext_vector_type(8)))  float  f32x8;

// ---- WMMA helper (CDNA5 wave32, v_wmma_f32_16x16x32_bf16) -------------------

static __device__ inline f32x8 wmma_bf16(bf16x16 a, bf16x16 b, f32x8 c) {
    return __builtin_amdgcn_wmma_f32_16x16x32_bf16(
        /*neg_a=*/false, a, /*neg_b=*/false, b,
        /*c_mod=*/(short)0, c, /*reuse_a=*/false, /*reuse_b=*/false);
}

// A-matrix fragment (16x32 bf16, ISA 7.12.2): lane L(<16): row L, K={0..7,16..23};
// lane L+16: row L, K={8..15,24..31}. base points at (row0,k0), row stride ld.
static __device__ inline bf16x16 load_afrag(const __bf16* base, int ld, int lane) {
    int row = lane & 15;
    int kh  = (lane >> 4) << 3;          // 0 or 8
    union { bf16x16 f; bf16x8 h[2]; } u;
    u.h[0] = *reinterpret_cast<const bf16x8*>(base + row * ld + kh);
    u.h[1] = *reinterpret_cast<const bf16x8*>(base + row * ld + kh + 16);
    return u.f;
}

// B-matrix fragment (32x16 bf16): lane L(<16): col L, K={0..7,16..23};
// lane L+16: col L, K={8..15,24..31}. base points at (k0,col0), row stride ld.
static __device__ inline bf16x16 load_bfrag(const __bf16* base, int ld, int lane) {
    int col = lane & 15;
    int kh  = (lane >> 4) << 3;
    bf16x16 r;
#pragma unroll
    for (int s = 0; s < 8; ++s) {
        r[s]     = base[(size_t)(kh + s) * ld + col];
        r[s + 8] = base[(size_t)(kh + 16 + s) * ld + col];
    }
    return r;
}

static __device__ inline float sigmoidf_(float x) { return 1.0f / (1.0f + __expf(-x)); }

// ---- One-time prep kernels --------------------------------------------------

// At[b][n][m] = (bf16) A[b][m][n]
__global__ __launch_bounds__(256) void kTransposeA(const float* __restrict__ A,
                                                   __bf16* __restrict__ At) {
    __shared__ float t[32][33];
    const int b  = blockIdx.z;
    const int n0 = blockIdx.x * 32;
    const int m0 = blockIdx.y * 32;
    const int tx = threadIdx.x, ty = threadIdx.y;   // block (32,8)
    const float* Ab = A + (size_t)b * Nn * Nn;
    __bf16*      Ob = At + (size_t)b * Nn * Nn;
#pragma unroll
    for (int i = 0; i < 32; i += 8)
        t[ty + i][tx] = Ab[(size_t)(m0 + ty + i) * Nn + (n0 + tx)];
    __syncthreads();
#pragma unroll
    for (int i = 0; i < 32; i += 8)
        Ob[(size_t)(n0 + ty + i) * Nn + (m0 + tx)] = (__bf16)t[tx][ty + i];
}

// WT[k*H + j] = (bf16) W[j*H + k]
__global__ __launch_bounds__(256) void kTransposeW(const float* __restrict__ W,
                                                   __bf16* __restrict__ WT) {
    int idx = blockIdx.x * 256 + threadIdx.x;
    if (idx < Hn * Hn) {
        int k = idx / Hn, j = idx % Hn;
        WT[idx] = (__bf16)W[j * Hn + k];
    }
}

// d_out (fp32 state) = hidden; hb0 (bf16 shadow) = hidden
__global__ __launch_bounds__(256) void kInitH(const float* __restrict__ hidden,
                                              float* __restrict__ hf,
                                              __bf16* __restrict__ hb, int n) {
    int i = blockIdx.x * 256 + threadIdx.x;
    if (i < n) {
        float v = hidden[i];
        hf[i] = v;
        hb[i] = (__bf16)v;
    }
}

// ---- Fused step kernel ------------------------------------------------------
// Block = 128 threads (4 waves), owns 64 rows (4 n-tiles) of one batch.
// Phase 0: a = At(64xN) @ hb_in(NxH) + b_ah          -> a_s (LDS, bf16)
// Phase 1: z,r gates (a_s,h_s A-frags; weights B-frags reused over 4 n-tiles)
// Phase 2: c = tanh(...); h' = (1-z)h + z c          -> hf (in place), hb_out
__global__ __launch_bounds__(128) void kStep(const __bf16* __restrict__ At,
                                             const __bf16* __restrict__ hb_in,
                                             __bf16* __restrict__ hb_out,
                                             float* __restrict__ hf,
                                             const float* __restrict__ b_ah,
                                             const __bf16* __restrict__ WT, // wz,uz,wr,ur,w,u
                                             const float* __restrict__ bwz,
                                             const float* __restrict__ buz,
                                             const float* __restrict__ bwr,
                                             const float* __restrict__ bur,
                                             const float* __restrict__ bw,
                                             const float* __restrict__ bu) {
    __shared__ __bf16 slab[32 * LDP];   // hb_in K-slab (32 x 128)
    __shared__ __bf16 a_s [64 * LDP];   // a tile
    __shared__ __bf16 h_s [64 * LDP];   // h tile (bf16)
    __shared__ __bf16 rh_s[64 * LDP];   // r*h tile

    const int n0   = blockIdx.x * 64;
    const int b    = blockIdx.y;
    const int tid  = threadIdx.x;
    const int lane = tid & 31;
    const int wv   = tid >> 5;

    const size_t bbase = (size_t)b * Nn * Hn;
    const size_t tbase = bbase + (size_t)n0 * Hn;

    // stage h tile (own 64 rows, contiguous 16 KiB) into LDS: 4 chunks / thread
    {
        const bf16x16* src = reinterpret_cast<const bf16x16*>(hb_in + tbase);
#pragma unroll
        for (int it = 0; it < 4; ++it) {
            int idx = tid + it * 128;            // 0..511, uniform trip count
            int row = idx >> 3, c = idx & 7;
            *reinterpret_cast<bf16x16*>(h_s + row * LDP + c * 16) = src[idx];
        }
    }

    const int col0  = lane & 15;
    const int rbase = (lane >> 4) * 8;

    // ---------- Phase 0: a = At @ h  (each wave: one 16-row tile, K = N) -----
    {
        const __bf16* Ab = At + (size_t)b * Nn * Nn + (size_t)(n0 + wv * 16) * Nn;
        f32x8 acc[8];
#pragma unroll
        for (int j = 0; j < 8; ++j) acc[j] = (f32x8){};

        for (int k0 = 0; k0 < Nn; k0 += 32) {
            // afrag global load issued early; in flight across barrier + stores
            bf16x16 af = load_afrag(Ab + k0, Nn, lane);
            __syncthreads();   // previous slab fully consumed (also covers h_s load)
            {
                const bf16x16* src =
                    reinterpret_cast<const bf16x16*>(hb_in + bbase + (size_t)k0 * Hn);
#pragma unroll
                for (int it = 0; it < 2; ++it) {
                    int idx = tid + it * 128;    // 0..255, uniform trip count
                    int row = idx >> 3, c = idx & 7;
                    *reinterpret_cast<bf16x16*>(slab + row * LDP + c * 16) = src[idx];
                }
            }
            __syncthreads();
            // preload all 8 B-fragments, then issue 8 WMMAs (hide LDS latency)
            bf16x16 bfv[8];
#pragma unroll
            for (int j = 0; j < 8; ++j)
                bfv[j] = load_bfrag(slab + j * 16, LDP, lane);
#pragma unroll
            for (int j = 0; j < 8; ++j)
                acc[j] = wmma_bf16(af, bfv[j], acc[j]);
        }
        // write a tile (+bias) into LDS as bf16
#pragma unroll
        for (int j = 0; j < 8; ++j) {
            int   col  = j * 16 + col0;
            float bias = b_ah[col];
#pragma unroll
            for (int v = 0; v < 8; ++v) {
                int row = wv * 16 + v + rbase;
                a_s[row * LDP + col] = (__bf16)(acc[j][v] + bias);
            }
        }
    }
    __syncthreads();

    const __bf16* WzT = WT;
    const __bf16* UzT = WT + 1 * Hn * Hn;
    const __bf16* WrT = WT + 2 * Hn * Hn;
    const __bf16* UrT = WT + 3 * Hn * Hn;
    const __bf16* WcT = WT + 4 * Hn * Hn;
    const __bf16* UcT = WT + 5 * Hn * Hn;

    f32x8 zg[8];   // z gate, [q*4 + nt]

    // ---------- Phase 1: z, r gates; build r*h tile ---------------------------
#pragma unroll
    for (int q = 0; q < 2; ++q) {
        const int jt = wv * 2 + q;
        const int j0 = jt * 16;
        f32x8 az[4], ar[4];
#pragma unroll
        for (int nt = 0; nt < 4; ++nt) { az[nt] = (f32x8){}; ar[nt] = (f32x8){}; }

#pragma unroll
        for (int k0 = 0; k0 < Hn; k0 += 32) {
            // weight B-fragments: loaded once, reused across 4 n-tiles
            bf16x16 bz = load_bfrag(WzT + (size_t)k0 * Hn + j0, Hn, lane);
            bf16x16 bZ = load_bfrag(UzT + (size_t)k0 * Hn + j0, Hn, lane);
            bf16x16 br = load_bfrag(WrT + (size_t)k0 * Hn + j0, Hn, lane);
            bf16x16 bR = load_bfrag(UrT + (size_t)k0 * Hn + j0, Hn, lane);
#pragma unroll
            for (int nt = 0; nt < 4; ++nt) {
                bf16x16 af = load_afrag(a_s + nt * 16 * LDP + k0, LDP, lane);
                bf16x16 hg = load_afrag(h_s + nt * 16 * LDP + k0, LDP, lane);
                az[nt] = wmma_bf16(af, bz, az[nt]);
                az[nt] = wmma_bf16(hg, bZ, az[nt]);
                ar[nt] = wmma_bf16(af, br, ar[nt]);
                ar[nt] = wmma_bf16(hg, bR, ar[nt]);
            }
        }
        const int   j   = j0 + col0;
        const float bz_ = bwz[j] + buz[j];
        const float br_ = bwr[j] + bur[j];
#pragma unroll
        for (int nt = 0; nt < 4; ++nt) {
#pragma unroll
            for (int v = 0; v < 8; ++v) {
                float z = sigmoidf_(az[nt][v] + bz_);
                float r = sigmoidf_(ar[nt][v] + br_);
                int   row  = nt * 16 + v + rbase;
                float hval = (float)h_s[row * LDP + j];
                rh_s[row * LDP + j] = (__bf16)(r * hval);
                zg[q * 4 + nt][v] = z;
            }
        }
    }
    __syncthreads();

    // ---------- Phase 2: candidate + recurrence -------------------------------
#pragma unroll
    for (int q = 0; q < 2; ++q) {
        const int jt = wv * 2 + q;
        const int j0 = jt * 16;
        f32x8 ac[4];
#pragma unroll
        for (int nt = 0; nt < 4; ++nt) ac[nt] = (f32x8){};

#pragma unroll
        for (int k0 = 0; k0 < Hn; k0 += 32) {
            bf16x16 bW = load_bfrag(WcT + (size_t)k0 * Hn + j0, Hn, lane);
            bf16x16 bU = load_bfrag(UcT + (size_t)k0 * Hn + j0, Hn, lane);
#pragma unroll
            for (int nt = 0; nt < 4; ++nt) {
                bf16x16 af  = load_afrag(a_s  + nt * 16 * LDP + k0, LDP, lane);
                bf16x16 rhf = load_afrag(rh_s + nt * 16 * LDP + k0, LDP, lane);
                ac[nt] = wmma_bf16(af,  bW, ac[nt]);
                ac[nt] = wmma_bf16(rhf, bU, ac[nt]);
            }
        }
        const int   j   = j0 + col0;
        const float bc_ = bw[j] + bu[j];
#pragma unroll
        for (int nt = 0; nt < 4; ++nt) {
#pragma unroll
            for (int v = 0; v < 8; ++v) {
                float  c    = tanhf(ac[nt][v] + bc_);
                int    row  = nt * 16 + v + rbase;
                size_t gi   = tbase + (size_t)row * Hn + j;
                float  hold = hf[gi];
                float  z    = zg[q * 4 + nt][v];
                float  hnew = (1.0f - z) * hold + z * c;
                hf[gi]     = hnew;
                hb_out[gi] = (__bf16)hnew;
            }
        }
    }
}

// ---- Launch -----------------------------------------------------------------

extern "C" void kernel_launch(void* const* d_in, const int* in_sizes, int n_in,
                              void* d_out, int out_size, void* d_ws, size_t ws_size,
                              hipStream_t stream) {
    const float* A      = (const float*)d_in[0];
    const float* hidden = (const float*)d_in[1];
    const float* b_ah   = (const float*)d_in[2];
    const float* w_z    = (const float*)d_in[3];
    const float* b_wz   = (const float*)d_in[4];
    const float* u_z    = (const float*)d_in[5];
    const float* b_uz   = (const float*)d_in[6];
    const float* w_r    = (const float*)d_in[7];
    const float* b_wr   = (const float*)d_in[8];
    const float* u_r    = (const float*)d_in[9];
    const float* b_ur   = (const float*)d_in[10];
    const float* w_     = (const float*)d_in[11];
    const float* b_w    = (const float*)d_in[12];
    const float* u_     = (const float*)d_in[13];
    const float* b_u    = (const float*)d_in[14];

    float* hf = (float*)d_out;   // fp32 hidden state, updated in place

    char* ws = (char*)d_ws;
    const size_t atB = (size_t)Bn * Nn * Nn * 2;   // 64 MiB
    const size_t hbB = (size_t)Bn * Nn * Hn * 2;   // 16 MiB
    __bf16* At  = (__bf16*)(ws);
    __bf16* hb0 = (__bf16*)(ws + atB);
    __bf16* hb1 = (__bf16*)(ws + atB + hbB);
    __bf16* WT  = (__bf16*)(ws + atB + 2 * hbB);   // 6 * 128*128 bf16

    // One-time prep
    kTransposeA<<<dim3(Nn / 32, Nn / 32, Bn), dim3(32, 8), 0, stream>>>(A, At);

    const float* wsrc[6] = {w_z, u_z, w_r, u_r, w_, u_};
    for (int i = 0; i < 6; ++i)
        kTransposeW<<<dim3((Hn * Hn + 255) / 256), dim3(256), 0, stream>>>(
            wsrc[i], WT + (size_t)i * Hn * Hn);

    const int nh = Bn * Nn * Hn;
    kInitH<<<dim3((nh + 255) / 256), dim3(256), 0, stream>>>(hidden, hf, hb0, nh);

    // T fused recurrence steps with bf16 h ping-pong
    __bf16* hbs[2] = {hb0, hb1};
    for (int t = 0; t < Tn; ++t) {
        kStep<<<dim3(Nn / 64, Bn), dim3(128), 0, stream>>>(
            At, hbs[t & 1], hbs[(t + 1) & 1], hf, b_ah, WT,
            b_wz, b_uz, b_wr, b_ur, b_w, b_u);
    }
}
